// BOW_30631706755077
// MI455X (gfx1250) — matrix operand order repaired and checked by reading, not compiled, not run on hardware
//
#include <hip/hip_runtime.h>

// Problem constants (from reference): B=16, S=64, L=1024, V=50000, D=512
#define BB 16
#define SS 64
#define LL 1024
#define VV 50000
#define DD 512

#define NT 32     // N-tile (out columns) per workgroup -> 16x16 = 256 WGs
#define KT 32     // K-step = bf16 WMMA K depth
#define PITCH 40  // padded K pitch (bf16 elems): 80B rows, 16B-aligned b128, bank-friendly

typedef __attribute__((ext_vector_type(16))) __bf16 v16bf;
typedef __attribute__((ext_vector_type(8)))  float  v8f;

// ---------------------------------------------------------------------------
// Kernel 1: prev[b][p] = last q < p with tok[b][q] == tok[b][p], else -1.
// ---------------------------------------------------------------------------
__global__ __launch_bounds__(256) void bow_prev_kernel(const int* __restrict__ ids,
                                                       int* __restrict__ prev) {
    const int b = blockIdx.x;
    __shared__ int toks[LL];
    for (int p = threadIdx.x; p < LL; p += 256) toks[p] = ids[b * LL + p];
    __syncthreads();
    for (int p = threadIdx.x; p < LL; p += 256) {
        const int t = toks[p];
        int pr = -1;
        for (int q = p - 1; q >= 0; --q) {
            if (toks[q] == t) { pr = q; break; }
        }
        prev[b * LL + p] = pr;
    }
}

// ---------------------------------------------------------------------------
// Kernel 2: out[b] = MASK[b] (64x1024, 0/1) x G[b] (1024x512) + bias,
// G[b][p,d] = W[d, tok[b,p]] gathered on the fly (L2-resident), bf16 WMMA with
// hi/lo split of W. grid = (B, D/32); 128 threads = 4 waves, wave w -> 16-row
// strip, 2 column subtiles. Gather is register-double-buffered one K-step ahead.
// ---------------------------------------------------------------------------
__global__ __launch_bounds__(128) void bow_gemm_kernel(const int*   __restrict__ ids,
                                                       const int*   __restrict__ spans,
                                                       const float* __restrict__ W,
                                                       const float* __restrict__ bias,
                                                       const int*   __restrict__ prev,
                                                       float*       __restrict__ out) {
    const int b        = blockIdx.x;
    const int ncolbase = blockIdx.y * NT;
    const int tid      = threadIdx.x;
    const int lane     = tid & 31;
    const int wv       = tid >> 5;  // wave id 0..3 -> row strip [16*wv, 16*wv+16)

    __shared__ int            tok_s[LL];
    __shared__ int            prev_s[LL];
    __shared__ unsigned short Bhi[NT * PITCH];
    __shared__ unsigned short Blo[NT * PITCH];

    // stage tokens + prev (vectorized, 512B/wave)
    for (int p = tid * 4; p < LL; p += 128 * 4) {
        *(int4*)&tok_s[p]  = *(const int4*)&ids[b * LL + p];
        *(int4*)&prev_s[p] = *(const int4*)&prev[b * LL + p];
    }

    // span bounds for this lane's output row
    const int srow = wv * 16 + (lane & 15);
    const int i0   = spans[((b * SS) + srow) * 2 + 0];
    const int j0   = spans[((b * SS) + srow) * 2 + 1];

    // fill-phase mapping: each thread owns column n_f and 8 CONSECUTIVE k values.
    // Uniform SGPR base (W slice for this WG) + 32-bit per-thread offsets:
    // slice spans NT*V = 1.6M floats, so off_base + tok < 2^21 -> SADDR+voffset path.
    const int    n_f      = tid & (NT - 1);   // 0..31
    const int    kk_f     = (tid >> 5) << 3;  // 0,8,16,24
    const float* __restrict__ Wtile = W + (size_t)ncolbase * VV;  // uniform
    const unsigned off_base = (unsigned)(n_f * VV);               // per-thread

    v8f acc[2] = {};  // two 16x16 fp32 column subtiles

    __syncthreads();

    // ---- gather tile k0 into 8 regs (8 independent scattered 4B loads) ----
    auto gather = [&](int k0, float* r) {
        if (k0 >= LL) return;
        const int4 t0 = *(const int4*)&tok_s[k0 + kk_f + 0];
        const int4 t1 = *(const int4*)&tok_s[k0 + kk_f + 4];
        const int tk[8] = { t0.x, t0.y, t0.z, t0.w, t1.x, t1.y, t1.z, t1.w };
#pragma unroll
        for (int i = 0; i < 8; ++i) r[i] = Wtile[off_base + (unsigned)tk[i]];
    };

    // ---- hi/lo bf16 split (RNE hi, truncated lo residual; err ~2^-16 rel)
    //      + vectorized LDS store (1 b128 per plane) ----
    auto convert_store = [&](const float* r) {
        union { unsigned short u[8]; uint4 q; } ph, pl;
#pragma unroll
        for (int i = 0; i < 8; ++i) {
            const unsigned int u   = __float_as_uint(r[i]);
            const unsigned int hi  = (u + 0x7FFFu + ((u >> 16) & 1u)) >> 16;
            const float        fhi = __uint_as_float(hi << 16);
            const float        lo  = r[i] - fhi;  // exact (Sterbenz)
            ph.u[i] = (unsigned short)hi;
            pl.u[i] = (unsigned short)(__float_as_uint(lo) >> 16);
        }
        *(uint4*)&Bhi[n_f * PITCH + kk_f] = ph.q;
        *(uint4*)&Blo[n_f * PITCH + kk_f] = pl.q;
    };

    // ---- A-mask build (vectorized prev reads) + B operand loads + 4 WMMAs ----
    auto process = [&](int k0) {
        // 16-bit A 16x32 layout: lanes 0-15 hold K {0..7,16..23}, lanes 16-31 {8..15,24..31}
        const int kbase = (lane < 16) ? 0 : 8;
        const int4 p0 = *(const int4*)&prev_s[k0 + kbase + 0];
        const int4 p1 = *(const int4*)&prev_s[k0 + kbase + 4];
        const int4 p2 = *(const int4*)&prev_s[k0 + kbase + 16];
        const int4 p3 = *(const int4*)&prev_s[k0 + kbase + 20];
        const int pv[16] = { p0.x, p0.y, p0.z, p0.w, p1.x, p1.y, p1.z, p1.w,
                             p2.x, p2.y, p2.z, p2.w, p3.x, p3.y, p3.z, p3.w };
        union { v16bf v; unsigned short e[16]; } A;
#pragma unroll
        for (int s = 0; s < 16; ++s) {
            const int K = kbase + ((s < 8) ? s : (s + 8));
            const int p = k0 + K;
            const bool on = (p >= i0) && (p < j0) && (pv[s] < i0);
            A.e[s] = on ? (unsigned short)0x3F80 : (unsigned short)0;  // bf16 1.0/0.0
        }

        const int nloc = lane & 15;
        const int krow = (lane < 16) ? 0 : 8;  // B 32x16: lanes 0-15 K={0..7,16..23}, 16-31 K={8..15,24..31}
#pragma unroll
        for (int c = 0; c < 2; ++c) {
            const int off = (c * 16 + nloc) * PITCH + krow;
            union { v16bf v; uint4 q[2]; } bh, bl;
            bh.q[0] = *(const uint4*)&Bhi[off];
            bh.q[1] = *(const uint4*)&Bhi[off + 16];
            bl.q[0] = *(const uint4*)&Blo[off];
            bl.q[1] = *(const uint4*)&Blo[off + 16];
            acc[c] = __builtin_amdgcn_wmma_f32_16x16x32_bf16(
                false, A.v, false, bh.v, (short)0, acc[c], false, false);
            acc[c] = __builtin_amdgcn_wmma_f32_16x16x32_bf16(
                false, A.v, false, bl.v, (short)0, acc[c], false, false);
        }
    };

    // ---- main loop, unrolled x2 to ping-pong register buffers ----
    float ra[8], rb[8];
    gather(0, ra);
    for (int k0 = 0; k0 < LL; k0 += 2 * KT) {
        gather(k0 + KT, rb);      // prefetch next tile (latency hidden past WMMA phase)
        convert_store(ra);
        __syncthreads();          // tile visible to all waves
        process(k0);
        __syncthreads();          // all waves done reading before next overwrite

        gather(k0 + 2 * KT, ra);
        convert_store(rb);
        __syncthreads();
        process(k0 + KT);
        __syncthreads();
    }

    // ---- epilogue: f32 C/D 16x16 layout: lanes 0-15 -> M=r, lanes 16-31 -> M=8+r.
    // Non-temporal stores: don't let the 2MB result stream evict L2-resident W.
    const int Nl    = lane & 15;
    const int mbase = (lane < 16) ? 0 : 8;
#pragma unroll
    for (int c = 0; c < 2; ++c) {
        const int   dcol = ncolbase + c * 16 + Nl;
        const float bv   = bias[dcol];
#pragma unroll
        for (int r = 0; r < 8; ++r) {
            const int srow2 = wv * 16 + mbase + r;
            __builtin_nontemporal_store(acc[c][r] + bv,
                                        &out[((size_t)b * SS + srow2) * DD + dcol]);
        }
    }
}

extern "C" void kernel_launch(void* const* d_in, const int* in_sizes, int n_in,
                              void* d_out, int out_size, void* d_ws, size_t ws_size,
                              hipStream_t stream) {
    const int*   ids   = (const int*)d_in[0];    // input_ids [B,L] int32
    const int*   spans = (const int*)d_in[1];    // span_idxs [B,S,2] int32
    const float* W     = (const float*)d_in[2];  // [D,V] fp32
    const float* bias  = (const float*)d_in[3];  // [D] fp32
    float*       out   = (float*)d_out;          // [B,S,D] fp32

    int* prev = (int*)d_ws;  // [B,L] int32 scratch (64 KB)

    bow_prev_kernel<<<BB, 256, 0, stream>>>(ids, prev);

    dim3 grid(BB, DD / NT);  // 16 x 16 = 256 workgroups
    bow_gemm_kernel<<<grid, 128, 0, stream>>>(ids, spans, W, bias, prev, out);
}
